// ExperimentalLayer9_1563368096372
// MI455X (gfx1250) — matrix operand order repaired and checked by reading, not compiled, not run on hardware
//
#include <hip/hip_runtime.h>
#include <stdint.h>

#define D_MODEL 1024
#define N_HEADS 16
#define D_HIDDEN 4096
#define BATCH 2
#define N_CTX 2048
#define DK (D_MODEL / N_HEADS)    // 64
#define DV (D_HIDDEN / N_HEADS)   // 256
#define ROWS (BATCH * N_CTX)      // 4096

typedef __attribute__((ext_vector_type(16))) __bf16 v16bf;
typedef __attribute__((ext_vector_type(8)))  float  v8f;

// ---------------------------------------------------------------- helpers
__device__ __forceinline__ uint16_t f2bf(float f) {
  uint32_t x = __float_as_uint(f);
  x += 0x7FFFu + ((x >> 16) & 1u);   // round-to-nearest-even
  return (uint16_t)(x >> 16);
}

__device__ __forceinline__ v8f wmma_bf16(v16bf a, v16bf b, v8f c) {
  // D = A(16x32 bf16) * B(32x16 bf16) + C(16x16 f32)
  return __builtin_amdgcn_wmma_f32_16x16x32_bf16(
      false, a, false, b, (short)0, c, false, false);
}

// A operand: 16x32 (MxK) tile from row-major bf16 matrix, leading dim `ld`.
// Lane layout (ISA 7.12.2): lanes 0-15 m=lane, K = {8*hi..8*hi+7, 16+8*hi..+7}
__device__ __forceinline__ v16bf load_fragA(const uint16_t* base, int ld,
                                            int row0, int col0) {
  int lane = threadIdx.x & 31;
  int m = lane & 15, hi = lane >> 4;
  const uint16_t* p = base + (size_t)(row0 + m) * ld + col0 + 8 * hi;
  union { v16bf v; uint4 q[2]; } f;
  f.q[0] = *(const uint4*)(p);
  f.q[1] = *(const uint4*)(p + 16);
  return f.v;
}

// A operand from LDS, fixed ld=32, tile origin 0
__device__ __forceinline__ v16bf load_fragA_lds(const uint16_t* base) {
  int lane = threadIdx.x & 31;
  int m = lane & 15, hi = lane >> 4;
  const uint16_t* p = base + m * 32 + 8 * hi;
  union { v16bf v; uint4 q[2]; } f;
  f.q[0] = *(const uint4*)(p);
  f.q[1] = *(const uint4*)(p + 16);
  return f.v;
}

// B operand: 32x16 (KxN), sourced from B^T stored row-major [N][K], ld = K dim.
// Lane n = lane&15 holds K = {8*hi..8*hi+7, 16+8*hi..+7} (same order as A).
__device__ __forceinline__ v16bf load_fragB(const uint16_t* base, int ld,
                                            int n0, int col0) {
  int lane = threadIdx.x & 31;
  int n = lane & 15, hi = lane >> 4;
  const uint16_t* p = base + (size_t)(n0 + n) * ld + col0 + 8 * hi;
  union { v16bf v; uint4 q[2]; } f;
  f.q[0] = *(const uint4*)(p);
  f.q[1] = *(const uint4*)(p + 16);
  return f.v;
}

__device__ __forceinline__ float redmax16(float v) {
#pragma unroll
  for (int m = 1; m < 16; m <<= 1) v = fmaxf(v, __shfl_xor(v, m, 32));
  return v;
}
__device__ __forceinline__ float redsum16(float v) {
#pragma unroll
  for (int m = 1; m < 16; m <<= 1) v += __shfl_xor(v, m, 32);
  return v;
}

// 32x64 output macro-tile per wave: 2 A-frags x 4 B-frags = 8 WMMA / k-step.
__device__ __forceinline__ void gemm_2x4(const uint16_t* A, int lda, int m0,
                                         const uint16_t* Bt, int ldb, int n0,
                                         int Kdim, v8f acc[2][4]) {
#pragma unroll
  for (int i = 0; i < 2; i++)
#pragma unroll
    for (int t = 0; t < 4; t++) acc[i][t] = (v8f){};
  for (int k = 0; k < Kdim; k += 32) {
    v16bf a0 = load_fragA(A, lda, m0, k);
    v16bf a1 = load_fragA(A, lda, m0 + 16, k);
#pragma unroll
    for (int t = 0; t < 4; t++) {
      v16bf b = load_fragB(Bt, ldb, n0 + 16 * t, k);
      acc[0][t] = wmma_bf16(a0, b, acc[0][t]);
      acc[1][t] = wmma_bf16(a1, b, acc[1][t]);
    }
  }
}

// ---------------------------------------------------------------- prep
__global__ __launch_bounds__(256) void cvt_f32_bf16(const float* __restrict__ in,
                                                    uint16_t* __restrict__ out,
                                                    int n) {
  int i = blockIdx.x * 256 + threadIdx.x;
  if (i < n) out[i] = f2bf(in[i]);
}

// W[K][N] fp32 -> Wt[N][K] bf16
__global__ __launch_bounds__(256) void transpose_bf(const float* __restrict__ W,
                                                    uint16_t* __restrict__ Wt,
                                                    int K, int N) {
  int i = blockIdx.x * 256 + threadIdx.x;
  if (i < K * N) {
    int k = i / N, n = i % N;
    Wt[(size_t)n * K + k] = f2bf(W[i]);
  }
}

// ---------------------------------------------------------------- K projection
__global__ __launch_bounds__(256)
void kproj_kernel(const uint16_t* __restrict__ xb,
                  const uint16_t* __restrict__ WkT,
                  const float* __restrict__ bk,
                  uint16_t* __restrict__ Kb) {
  int wave = (blockIdx.x * blockDim.x + threadIdx.x) >> 5;
  int lane = threadIdx.x & 31;
  const int ng = D_MODEL / 64;  // 16
  int m0 = (wave / ng) * 32;
  int n0 = (wave % ng) * 64;
  v8f acc[2][4];
  gemm_2x4(xb, D_MODEL, m0, WkT, D_MODEL, n0, D_MODEL, acc);
  int hi = lane >> 4;
#pragma unroll
  for (int i = 0; i < 2; i++)
#pragma unroll
    for (int t = 0; t < 4; t++) {
      int n = n0 + 16 * t + (lane & 15);
      float bias = bk[n];
#pragma unroll
      for (int r = 0; r < 8; r++) {
        int row = m0 + 16 * i + r + 8 * hi;
        Kb[(size_t)row * D_MODEL + n] = f2bf(acc[i][t][r] + bias);
      }
    }
}

// ---------------------------------------------------------------- V projection
__global__ __launch_bounds__(256)
void vproj_kernel(const uint16_t* __restrict__ xb,
                  const uint16_t* __restrict__ WvT,
                  const float* __restrict__ bv,
                  uint16_t* __restrict__ Vt) {
  int wave = (blockIdx.x * blockDim.x + threadIdx.x) >> 5;
  int lane = threadIdx.x & 31;
  const int ng = D_HIDDEN / 64;  // 64
  int m0 = (wave / ng) * 32;
  int n0 = (wave % ng) * 64;
  v8f acc[2][4];
  gemm_2x4(xb, D_MODEL, m0, WvT, D_MODEL, n0, D_MODEL, acc);
  int hi = lane >> 4;
#pragma unroll
  for (int i = 0; i < 2; i++)
#pragma unroll
    for (int t = 0; t < 4; t++) {
      int n = n0 + 16 * t + (lane & 15);
      float bias = bv[n];
      int h = n >> 8, d = n & 255;
#pragma unroll
      for (int r = 0; r < 8; r++) {
        int row = m0 + 16 * i + r + 8 * hi;
        int b = row >> 11, s = row & (N_CTX - 1);
        // Vt[b][h][d][s]  (== (V^T)[d][s] per head: B^T layout for P*V)
        Vt[((((size_t)b * N_HEADS + h) * DV) + d) * N_CTX + s] =
            f2bf(acc[i][t][r] + bias);
      }
    }
}

// ---------------------------------------------------------------- attention
__global__ __launch_bounds__(256)
void attn_kernel(const uint16_t* __restrict__ xb,   // Q source (bf16 x)
                 const uint16_t* __restrict__ Kb,   // [ROWS][D_MODEL]
                 const uint16_t* __restrict__ Vt,   // [B][H][DV][N_CTX]
                 uint16_t* __restrict__ g) {        // gelu(o) [ROWS][D_HIDDEN]
  __shared__ __align__(16) uint16_t Plds[8][16 * 32];
  int waveid = threadIdx.x >> 5;
  int lane = threadIdx.x & 31;
  int tile = blockIdx.x * 8 + waveid;       // B*H*(S/16) = 4096 tiles
  int qt = tile & 127;
  int h = (tile >> 7) & 15;
  int b = tile >> 11;
  int q0 = qt * 16;

  const uint16_t* Qbase = xb + (size_t)b * N_CTX * D_MODEL + h * DK;
  const uint16_t* Kbase = Kb + (size_t)b * N_CTX * D_MODEL + h * DK;
  const uint16_t* Vbase = Vt + (((size_t)b * N_HEADS + h) * DV) * N_CTX;
  uint16_t* P = &Plds[waveid][0];

  v16bf qa0 = load_fragA(Qbase, D_MODEL, q0, 0);
  v16bf qa1 = load_fragA(Qbase, D_MODEL, q0, 32);

  v8f acc[16];
#pragma unroll
  for (int t = 0; t < 16; t++) acc[t] = (v8f){};
  float mrun[8], lrun[8];
#pragma unroll
  for (int r = 0; r < 8; r++) { mrun[r] = -INFINITY; lrun[r] = 0.0f; }

  int n = lane & 15, hi = lane >> 4;
  int nchunk = (q0 + 16 + 31) >> 5;  // keys [0, q0+16) in chunks of 32

  for (int kc = 0; kc < nchunk; ++kc) {
    int kk0 = kc << 5;
    // ---- scores: 16 q-rows x 32 keys (two 16x16 tiles), reduce over d=64
    v8f s0 = {};
    s0 = wmma_bf16(qa0, load_fragB(Kbase, D_MODEL, kk0, 0), s0);
    s0 = wmma_bf16(qa1, load_fragB(Kbase, D_MODEL, kk0, 32), s0);
    v8f s1 = {};
    s1 = wmma_bf16(qa0, load_fragB(Kbase, D_MODEL, kk0 + 16, 0), s1);
    s1 = wmma_bf16(qa1, load_fragB(Kbase, D_MODEL, kk0 + 16, 32), s1);

    int col0 = kk0 + n, col1 = kk0 + 16 + n;
    float al8[8];
#pragma unroll
    for (int r = 0; r < 8; r++) {
      int row = q0 + r + 8 * hi;
      float v0 = (col0 <= row) ? s0[r] : -INFINITY;  // causal mask
      float v1 = (col1 <= row) ? s1[r] : -INFINITY;
      float rm = redmax16(fmaxf(v0, v1));
      float mn = fmaxf(mrun[r], rm);
      float al = __expf(mrun[r] - mn);               // first iter: exp(-inf)=0
      float p0 = __expf(v0 - mn);
      float p1 = __expf(v1 - mn);
      lrun[r] = lrun[r] * al + redsum16(p0 + p1);
      mrun[r] = mn;
      al8[r] = al;
      P[(r + 8 * hi) * 32 + n]      = f2bf(p0);
      P[(r + 8 * hi) * 32 + 16 + n] = f2bf(p1);
    }
#pragma unroll
    for (int t = 0; t < 16; t++)
#pragma unroll
      for (int r = 0; r < 8; r++) acc[t][r] *= al8[r];

    asm volatile("s_wait_dscnt 0" ::: "memory");  // cross-lane LDS fence
    v16bf pa = load_fragA_lds(P);
#pragma unroll
    for (int t = 0; t < 16; t++) {
      v16bf vb = load_fragB(Vbase, N_CTX, t * 16, kk0);
      acc[t] = wmma_bf16(pa, vb, acc[t]);
    }
  }

  // ---- normalize, exact GELU, store bf16
  float inv8[8];
#pragma unroll
  for (int r = 0; r < 8; r++) inv8[r] = 1.0f / lrun[r];
#pragma unroll
  for (int t = 0; t < 16; t++) {
#pragma unroll
    for (int r = 0; r < 8; r++) {
      float o = acc[t][r] * inv8[r];
      float ge = 0.5f * o * (1.0f + erff(o * 0.70710678118654752f));
      int row = b * N_CTX + q0 + r + 8 * hi;
      int col = h * DV + t * 16 + n;
      g[(size_t)row * D_HIDDEN + col] = f2bf(ge);
    }
  }
}

// ---------------------------------------------------------------- FFN out + residual
__global__ __launch_bounds__(256)
void ffn_kernel(const uint16_t* __restrict__ g,
                const uint16_t* __restrict__ WfT,
                const float* __restrict__ bf_,
                const float* __restrict__ x,
                float* __restrict__ out) {
  int wave = (blockIdx.x * blockDim.x + threadIdx.x) >> 5;
  int lane = threadIdx.x & 31;
  const int ng = D_MODEL / 64;  // 16
  int m0 = (wave / ng) * 32;
  int n0 = (wave % ng) * 64;
  v8f acc[2][4];
  gemm_2x4(g, D_HIDDEN, m0, WfT, D_HIDDEN, n0, D_HIDDEN, acc);
  int hi = lane >> 4;
#pragma unroll
  for (int i = 0; i < 2; i++)
#pragma unroll
    for (int t = 0; t < 4; t++) {
      int n = n0 + 16 * t + (lane & 15);
      float bias = bf_[n];
#pragma unroll
      for (int r = 0; r < 8; r++) {
        size_t idx = (size_t)(m0 + 16 * i + r + 8 * hi) * D_MODEL + n;
        out[idx] = x[idx] + acc[i][t][r] + bias;
      }
    }
}

// ---------------------------------------------------------------- launch
extern "C" void kernel_launch(void* const* d_in, const int* in_sizes, int n_in,
                              void* d_out, int out_size, void* d_ws, size_t ws_size,
                              hipStream_t stream) {
  const float* x  = (const float*)d_in[0];
  const float* Wk = (const float*)d_in[1];
  const float* bk = (const float*)d_in[2];
  const float* Wv = (const float*)d_in[3];
  const float* bv = (const float*)d_in[4];
  const float* Wf = (const float*)d_in[5];
  const float* bf_ = (const float*)d_in[6];
  float* out = (float*)d_out;

  uint8_t* w = (uint8_t*)d_ws;
  size_t off = 0;
  uint16_t* xb  = (uint16_t*)(w + off); off += (size_t)ROWS * D_MODEL * 2;       // 8 MB
  uint16_t* Kb  = (uint16_t*)(w + off); off += (size_t)ROWS * D_MODEL * 2;       // 8 MB
  uint16_t* WkT = (uint16_t*)(w + off); off += (size_t)D_MODEL * D_MODEL * 2;    // 2 MB
  uint16_t* WvT = (uint16_t*)(w + off); off += (size_t)D_HIDDEN * D_MODEL * 2;   // 8 MB
  uint16_t* WfT = (uint16_t*)(w + off); off += (size_t)D_MODEL * D_HIDDEN * 2;   // 8 MB
  uint16_t* Vt  = (uint16_t*)(w + off); off += (size_t)ROWS * D_HIDDEN * 2;      // 32 MB
  uint16_t* g   = (uint16_t*)(w + off); off += (size_t)ROWS * D_HIDDEN * 2;      // 32 MB
  (void)ws_size; (void)in_sizes; (void)n_in; (void)out_size;

  // prep: bf16 conversions / weight transposes
  cvt_f32_bf16<<<(ROWS * D_MODEL + 255) / 256, 256, 0, stream>>>(x, xb, ROWS * D_MODEL);
  transpose_bf<<<(D_MODEL * D_MODEL + 255) / 256, 256, 0, stream>>>(Wk, WkT, D_MODEL, D_MODEL);
  transpose_bf<<<(D_MODEL * D_HIDDEN + 255) / 256, 256, 0, stream>>>(Wv, WvT, D_MODEL, D_HIDDEN);
  transpose_bf<<<(D_HIDDEN * D_MODEL + 255) / 256, 256, 0, stream>>>(Wf, WfT, D_HIDDEN, D_MODEL);

  // projections: 1 wave = one 32x64 output macro-tile, 8 waves/block
  kproj_kernel<<<(ROWS / 32) * (D_MODEL / 64) / 8, 256, 0, stream>>>(xb, WkT, bk, Kb);
  vproj_kernel<<<(ROWS / 32) * (D_HIDDEN / 64) / 8, 256, 0, stream>>>(xb, WvT, bv, Vt);

  // causal flash attention + fused exact GELU (1 wave per 16-row q tile)
  attn_kernel<<<(BATCH * N_HEADS * (N_CTX / 16)) / 8, 256, 0, stream>>>(xb, Kb, Vt, g);

  // output GEMM + bias + residual
  ffn_kernel<<<(ROWS / 32) * (D_MODEL / 64) / 8, 256, 0, stream>>>(g, WfT, bf_, x, out);
}
